// ChannelWiseAttention_39633958208134
// MI455X (gfx1250) — matrix operand order repaired and checked by reading, not compiled, not run on hardware
//
#include <hip/hip_runtime.h>
#include <hip/hip_bf16.h>
#include <math.h>

typedef __attribute__((ext_vector_type(16))) __bf16          v16bf;
typedef __attribute__((ext_vector_type(8)))  float           v8f;
typedef __attribute__((ext_vector_type(16))) unsigned short  v16u16;
typedef __attribute__((ext_vector_type(8)))  unsigned short  v8u16;

// Problem constants
#define BATCH   16
#define CTX     4096
#define DMODEL  384
#define NHEADS  6
#define HDIM    64
#define QKVC    1152          // 3*DMODEL
#define NTOK    (BATCH*CTX)   // 65536
#define KCH     409

#define NTILES  72            // 1152 / 16
#define KTILES  12            // 384 / 32

// LDS strides (padded, keep 16B/32B alignment)
#define XS  400               // ushort stride for x tile row (800B, mult of 16)
#define QS  1160              // float stride for qkv tile row

__device__ __forceinline__ unsigned short f32_to_bf16_rne(float f) {
    unsigned u = __builtin_bit_cast(unsigned, f);
    u += 0x7fffu + ((u >> 16) & 1u);          // round-to-nearest-even
    return (unsigned short)(u >> 16);
}

// ---------------------------------------------------------------------------
// Kernel 1: convert W (1152x384 f32) to bf16, repacked FRAGMENT-MAJOR:
//   frag (nt, kt) occupies 1024B: lane L's 32B at Wb2 + (nt*12+kt)*1024 + L*32
//   lane L covers column n = nt*16 + (L&15), k = kt*32 + 16*(L>>4) + i, i=0..15
// This makes every B-fragment load in the GEMM a base+imm global_load_b128.
// Result (~884KB) stays resident in L2 for the whole dispatch.
// ---------------------------------------------------------------------------
__global__ __launch_bounds__(256) void convert_w_kernel(
    const float* __restrict__ W, unsigned short* __restrict__ Wb2) {
    const int t = blockIdx.x * 256 + threadIdx.x;   // one (nt, kt, lane) triple
    if (t >= NTILES * KTILES * 32) return;
    const int lane = t & 31;
    const int kt   = (t >> 5) % KTILES;
    const int nt   = (t >> 5) / KTILES;
    const int n = nt * 16 + (lane & 15);
    const int kbase = kt * 32 + (lane >> 4) * 16;
    unsigned short* dst = Wb2 + (size_t)t * 16;
    const float* src = W + (size_t)n * DMODEL + kbase;
#pragma unroll
    for (int i = 0; i < 16; ++i) dst[i] = f32_to_bf16_rne(src[i]);
}

// ---------------------------------------------------------------------------
// Kernel 2: fused qkv GEMM (bf16 WMMA, f32 accum) + per-token head argmax
// attention + importance. One block = 16 tokens x full 1152 qkv columns.
// 8 waves; wave w owns N-tiles nt = w + 8*j, j = 0..8  (8*9 = 72).
// ---------------------------------------------------------------------------
__global__ __launch_bounds__(256) void qkv_attn_kernel(
    const float* __restrict__ x,
    const unsigned short* __restrict__ Wb2,      // fragment-major bf16 W
    const float* __restrict__ bias,
    float* __restrict__ outbuf,                  // [NTOK, 384]
    float* __restrict__ imp)                     // [NTOK]
{
    __shared__ unsigned short ax[16 * XS];   // 16x384 bf16 x-tile (12.8 KB)
    __shared__ float qkv[16 * QS];           // 16x1152 f32 qkv tile (74.2 KB)
    __shared__ int   gstar[16 * NHEADS];
    __shared__ float hsum[16 * NHEADS];

    const int tid  = threadIdx.x;
    const int wave = tid >> 5;
    const int lane = tid & 31;
    const int half = lane >> 4;              // wave32: lanes 0-15 / 16-31
    const int l16  = lane & 15;
    const long tok0 = (long)blockIdx.x * 16;

    // --- stage x tile into LDS as bf16 (streamed once -> non-temporal) ----
    for (int e = tid; e < 16 * DMODEL; e += 256) {
        int r = e / DMODEL, c = e % DMODEL;
        float v = __builtin_nontemporal_load(&x[(tok0 + r) * DMODEL + c]);
        ax[r * XS + c] = f32_to_bf16_rne(v);
    }
    __syncthreads();

    // per-thread B base: all 108 fragment loads are base + immediate offset
    // byte addr = Wb2 + (nt*12+kt)*1024 + lane*32,  nt = wave + 8*j
    //           = [Wb2 + wave*12288 + lane*32] + j*98304 + kt*1024
    const unsigned short* bbase =
        (const unsigned short*)((const char*)Wb2 + wave * 12288 + lane * 32);

    v8f acc[9];
#pragma unroll
    for (int j = 0; j < 9; ++j) acc[j] = (v8f){0,0,0,0,0,0,0,0};

    for (int kt = 0; kt < KTILES; ++kt) {
        // A fragment: contiguous k order (matches B packing) -> per lane
        //   k(i) = 32*kt + 16*half + i, row M = l16
        const unsigned short* ap = &ax[l16 * XS + kt * 32 + half * 16];
        v8u16 alo = *(const v8u16*)(ap);
        v8u16 ahi = *(const v8u16*)(ap + 8);
        v16u16 au;
#pragma unroll
        for (int i = 0; i < 8; ++i) { au[i] = alo[i]; au[i + 8] = ahi[i]; }
        v16bf afrag = __builtin_bit_cast(v16bf, au);

#pragma unroll
        for (int j = 0; j < 9; ++j) {
            const v16u16 bu = *(const v16u16*)
                ((const char*)bbase + j * 98304 + kt * 1024);
            v16bf bfrag = __builtin_bit_cast(v16bf, bu);
            acc[j] = __builtin_amdgcn_wmma_f32_16x16x32_bf16(
                false, afrag, false, bfrag, (short)0, acc[j], false, false);
        }
    }

    // --- bias add + scatter C tiles to LDS qkv ----------------------------
    // C/D layout: VGPR r, lane -> M = r + 8*half, N = l16
#pragma unroll
    for (int j = 0; j < 9; ++j) {
        const int n = (wave + 8 * j) * 16 + l16;
        const float bv = bias[n];
#pragma unroll
        for (int r = 0; r < 8; ++r)
            qkv[(r + 8 * half) * QS + n] = acc[j][r] + bv;
    }
    __syncthreads();

    // --- per-(token, head): scores 1x6, top-1 -> source head g* ----------
    if (tid < 16 * NHEADS) {
        const int nl = tid / NHEADS, h = tid % NHEADS;
        const float* qp = &qkv[nl * QS + h * HDIM];
        float s[NHEADS] = {0, 0, 0, 0, 0, 0};
        for (int d = 0; d < HDIM; ++d) {
            const float qd = qp[d];
#pragma unroll
            for (int g = 0; g < NHEADS; ++g)
                s[g] += qd * qkv[nl * QS + DMODEL + g * HDIM + d];
        }
        int best = 0; float bv = s[0];
#pragma unroll
        for (int g = 1; g < NHEADS; ++g)
            if (s[g] > bv) { bv = s[g]; best = g; }   // strict > : tie -> lower idx
        gstar[tid] = best;
        // deterministic per-head sum of squares of selected v row
        const float* vp = &qkv[nl * QS + 2 * DMODEL + best * HDIM];
        float ss = 0.f;
        for (int d = 0; d < HDIM; ++d) ss += vp[d] * vp[d];
        hsum[tid] = ss;
    }
    __syncthreads();

    // --- importance (deterministic 6-way sum) -----------------------------
    if (tid < 16) {
        float t = 0.f;
#pragma unroll
        for (int h = 0; h < NHEADS; ++h) t += hsum[tid * NHEADS + h];
        imp[tok0 + tid] = sqrtf(t);
    }

    // --- write out[n, h*64+d] = v[n, g*, d] -------------------------------
    for (int e = tid; e < 16 * DMODEL; e += 256) {
        const int nl = e / DMODEL, d = e % DMODEL;
        const int g = gstar[nl * NHEADS + (d >> 6)];
        outbuf[(tok0 + nl) * DMODEL + d] =
            qkv[nl * QS + 2 * DMODEL + g * HDIM + (d & 63)];
    }
}

// ---------------------------------------------------------------------------
// Kernel 3: per-batch-row top-409 of 4096 via in-LDS bitonic sort of packed
// keys (value desc, index asc on ties -> matches jax.lax.top_k), then gather.
// ---------------------------------------------------------------------------
__global__ __launch_bounds__(256) void topk_gather_kernel(
    const float* __restrict__ imp,
    const float* __restrict__ outbuf,
    float* __restrict__ feat_out,        // [BATCH, KCH, DMODEL]
    float* __restrict__ idx_out)         // [BATCH, KCH] (as float)
{
    __shared__ unsigned long long keys[CTX];   // 32 KB
    const int b = blockIdx.x, tid = threadIdx.x;

    for (int c = tid; c < CTX; c += 256) {
        // importance >= 0 so f32 bits are order-preserving as unsigned
        unsigned vb = __builtin_bit_cast(unsigned, imp[(long)b * CTX + c]);
        keys[c] = ((unsigned long long)vb << 32) | (unsigned)(~c);
    }
    __syncthreads();

    for (int k = 2; k <= CTX; k <<= 1) {
        for (int j = k >> 1; j > 0; j >>= 1) {
            for (int i = tid; i < CTX; i += 256) {
                const int ixj = i ^ j;
                if (ixj > i) {
                    const bool desc = ((i & k) == 0);  // sort descending overall
                    unsigned long long a = keys[i], c = keys[ixj];
                    if (desc ? (a < c) : (a > c)) { keys[i] = c; keys[ixj] = a; }
                }
            }
            __syncthreads();
        }
    }

    for (int kk = tid; kk < KCH; kk += 256) {
        const int c = (int)(~(unsigned)keys[kk]);
        idx_out[b * KCH + kk] = (float)c;
    }
    for (int e = tid; e < KCH * DMODEL; e += 256) {
        const int kk = e / DMODEL, d = e % DMODEL;
        const int c = (int)(~(unsigned)keys[kk]);
        feat_out[((long)b * KCH + kk) * DMODEL + d] =
            outbuf[((long)b * CTX + c) * DMODEL + d];
    }
}

// ---------------------------------------------------------------------------
extern "C" void kernel_launch(void* const* d_in, const int* in_sizes, int n_in,
                              void* d_out, int out_size, void* d_ws, size_t ws_size,
                              hipStream_t stream) {
    const float* x    = (const float*)d_in[0];   // (16,4096,384)
    const float* W    = (const float*)d_in[1];   // (1152,384)
    const float* bias = (const float*)d_in[2];   // (1152,)

    char* ws = (char*)d_ws;
    unsigned short* Wb2  = (unsigned short*)ws;                         // 884,736 B
    float* outbuf        = (float*)(ws + (1u << 20));                   // 100,663,296 B
    float* imp           = (float*)(ws + (1u << 20) + 100663296u);      // 262,144 B

    float* feat = (float*)d_out;                         // 16*409*384 floats
    float* idxo = feat + (size_t)BATCH * KCH * DMODEL;   // 16*409 entries

    const int nfrag = NTILES * KTILES * 32;              // 27648 (nt,kt,lane)
    convert_w_kernel<<<(nfrag + 255) / 256, 256, 0, stream>>>(W, Wb2);
    qkv_attn_kernel<<<NTOK / 16, 256, 0, stream>>>(x, Wb2, bias, outbuf, imp);
    topk_gather_kernel<<<BATCH, 256, 0, stream>>>(imp, outbuf, feat, idxo);
}
